// ContrastiveLoss_28956669510249
// MI455X (gfx1250) — matrix operand order repaired and checked by reading, not compiled, not run on hardware
//
#include <hip/hip_runtime.h>
#include <hip/hip_bf16.h>

// ContrastiveLoss for MI455X (gfx1250, wave32):
//   l = mean_i[ -log( sum_j exp( (z_i/T)·(z_j/T) ) / N ) ]
// Fused GEMM -> exp -> row-sum. z (8MB) is L2-resident, so the kernel is
// compute bound on v_wmma_f32_16x16x32_f16 + v_exp_f32. fp32-class accuracy
// at f16 matrix rate via hi/lo error-compensated split:
//   a = Ah*Bh + Ah*Bl + Al*Bh  (e*e' term ~2^-22, negligible)
// Software-pipelined, double-buffered LDS staging: one barrier/iteration,
// next tile's global loads overlap the current tile's WMMA+exp.

typedef __attribute__((ext_vector_type(16))) _Float16 v16h;
typedef __attribute__((ext_vector_type(8)))  float    v8f;

#define N_ROWS     16384
#define DIM        128
#define INV_T      10.0f                   // 1/TEMPERATURE
#define LOG2E      1.4426950408889634f
#define LOG_N      9.704060527839234f      // log(16384) = 14*ln2
#define LDS_STRIDE 136                     // halves per B row (pad vs bank conflicts)
#define JSPLIT     4                       // column-range split for occupancy
#define JCHUNK     (N_ROWS / JSPLIT)       // 4096 columns per block
#define NTHREADS   512                     // 16 waves, 256 rows per block

__global__ __launch_bounds__(NTHREADS)
void cl_gemm_exp_rows(const float* __restrict__ z, float* __restrict__ rowsum) {
    __shared__ _Float16 lds_hi[2][16 * LDS_STRIDE];
    __shared__ _Float16 lds_lo[2][16 * LDS_STRIDE];

    const int tid  = threadIdx.x;
    const int wave = tid >> 5;           // 0..15
    const int lane = tid & 31;
    const int g    = lane >> 4;          // half-wave group (K-offset select)
    const int ln   = lane & 15;
    const int i0   = blockIdx.x * 256;   // 16 waves * 16 rows per block
    const int arow = i0 + wave * 16 + ln;

    // ---- A tile (16 rows x 128 K) -> registers in ISA f16 16x32 A-layout ----
    // lanes 0-15: elems 0-7 = K[8g .. 8g+7], elems 8-15 = K[8g+16 .. 8g+23]
    v16h a_hi[4], a_lo[4];
    #pragma unroll
    for (int c = 0; c < 4; ++c) {
        const int koff = g * 8 + 32 * c;
        #pragma unroll
        for (int e = 0; e < 16; ++e) {
            const int k = (e < 8) ? (koff + e) : (koff + 16 + (e - 8));
            float x = z[arow * DIM + k] * INV_T;
            _Float16 h = (_Float16)x;
            a_hi[c][e] = h;
            a_lo[c][e] = (_Float16)(x - (float)h);
        }
    }

    float acc[8];
    #pragma unroll
    for (int r = 0; r < 8; ++r) acc[r] = 0.0f;

    const int jbeg   = blockIdx.y * JCHUNK;
    const int ntiles = JCHUNK / 16;

    // this thread's 4 staged elements: f = tid + k*512 -> (row n, col kk) of tile
    float st[4];
    #pragma unroll
    for (int k = 0; k < 4; ++k) {
        const int f = tid + k * NTHREADS;
        st[k] = z[(jbeg + (f >> 7)) * DIM + (f & 127)] * INV_T;
    }

    for (int it = 0; it < ntiles; ++it) {
        const int buf = it & 1;
        // ---- convert + store the pre-loaded tile into LDS hi/lo ----
        #pragma unroll
        for (int k = 0; k < 4; ++k) {
            const int f  = tid + k * NTHREADS;
            const int n  = f >> 7;
            const int kk = f & 127;
            _Float16 h = (_Float16)st[k];
            lds_hi[buf][n * LDS_STRIDE + kk] = h;
            lds_lo[buf][n * LDS_STRIDE + kk] = (_Float16)(st[k] - (float)h);
        }
        __syncthreads();   // staging visible; also fences buffer reuse (see note)

        // ---- issue next tile's global loads; they overlap the WMMAs below ----
        if (it + 1 < ntiles) {
            const int jn = jbeg + (it + 1) * 16;
            #pragma unroll
            for (int k = 0; k < 4; ++k) {
                const int f = tid + k * NTHREADS;
                st[k] = z[(jn + (f >> 7)) * DIM + (f & 127)] * INV_T;
            }
            if (it + 2 < ntiles)   // warm L2/WGP$ two tiles ahead
                __builtin_prefetch(z + (size_t)(jbeg + (it + 2) * 16) * DIM + tid * 4, 0, 0);
        }

        // ---- 16x16 tile: C = Ah*Bh + Ah*Bl + Al*Bh over K=128 ----
        v8f c8 = {};
        #pragma unroll
        for (int c = 0; c < 4; ++c) {
            const int kbase = g * 8 + 32 * c;   // B layout: lane=col, elems=K[8g..8g+7],K[8g+16..8g+23]
            v16h b_hi, b_lo;
            #pragma unroll
            for (int e = 0; e < 16; ++e) {
                b_hi[e] = lds_hi[buf][ln * LDS_STRIDE + kbase + e + ((e >> 3) << 3)];
                b_lo[e] = lds_lo[buf][ln * LDS_STRIDE + kbase + e + ((e >> 3) << 3)];
            }
            c8 = __builtin_amdgcn_wmma_f32_16x16x32_f16(false, a_hi[c], false, b_hi,
                                                        (short)0, c8, false, false);
            c8 = __builtin_amdgcn_wmma_f32_16x16x32_f16(false, a_hi[c], false, b_lo,
                                                        (short)0, c8, false, false);
            c8 = __builtin_amdgcn_wmma_f32_16x16x32_f16(false, a_lo[c], false, b_hi,
                                                        (short)0, c8, false, false);
        }
        // exp of every tile element, accumulate per row (C-layout: VGPR r = row r+8g)
        #pragma unroll
        for (int r = 0; r < 8; ++r)
            acc[r] += __builtin_amdgcn_exp2f(c8[r] * LOG2E);
        // Note: no second barrier needed. Before any wave can overwrite buffer
        // b at iteration it+2, it must pass the barrier of iteration it+1,
        // which globally orders it after every wave's reads of buffer b at
        // iteration it.
    }

    // reduce each acc[r] across the 16 lanes of each half-wave group
    #pragma unroll
    for (int r = 0; r < 8; ++r) {
        float s = acc[r];
        #pragma unroll
        for (int m = 1; m < 16; m <<= 1)
            s += __shfl_xor(s, m, 32);
        if (ln == 0)
            rowsum[(size_t)blockIdx.y * N_ROWS + i0 + wave * 16 + r + 8 * g] = s;
    }
}

__global__ __launch_bounds__(256)
void cl_reduce(const float* __restrict__ rowsum, float* __restrict__ out) {
    __shared__ float red[256];
    const int tid = threadIdx.x;
    float s = 0.0f;
    for (int i = tid; i < N_ROWS; i += 256) {
        float srow = 0.0f;
        #pragma unroll
        for (int y = 0; y < JSPLIT; ++y)
            srow += rowsum[(size_t)y * N_ROWS + i];
        s += LOG_N - __logf(srow);     // -log(srow / N)
    }
    red[tid] = s;
    __syncthreads();
    for (int stride = 128; stride > 0; stride >>= 1) {
        if (tid < stride) red[tid] += red[tid + stride];
        __syncthreads();
    }
    if (tid == 0) out[0] = red[0] / (float)N_ROWS;
}

extern "C" void kernel_launch(void* const* d_in, const int* in_sizes, int n_in,
                              void* d_out, int out_size, void* d_ws, size_t ws_size,
                              hipStream_t stream) {
    (void)in_sizes; (void)n_in; (void)out_size; (void)ws_size;
    const float* z = (const float*)d_in[0];
    float* rowsum  = (float*)d_ws;            // JSPLIT * 16384 floats = 256 KB

    dim3 grid1(N_ROWS / 256, JSPLIT);         // 256 workgroups, 16 waves each
    cl_gemm_exp_rows<<<grid1, dim3(NTHREADS), 0, stream>>>(z, rowsum);
    cl_reduce<<<dim3(1), dim3(256), 0, stream>>>(rowsum, (float*)d_out);
}
